// PreActBlock_conv_Q_86517821211489
// MI455X (gfx1250) — 8-bit matrix operand order repaired and checked by reading, compiled, not run on hardware
//
#include <hip/hip_runtime.h>
#include <hip/hip_bf16.h>

typedef int v8i __attribute__((ext_vector_type(8)));

#define C_CH   256
#define H_DIM  56
#define W_DIM  56
#define HW     (H_DIM * W_DIM)
#define NIMG   16
#define ACT_ELEMS (NIMG * C_CH * HW)
#define W_ELEMS   (C_CH * C_CH * 9)
#define BPX    68

__global__ __launch_bounds__(256)
void wmax_kernel(const float* __restrict__ w, unsigned int* __restrict__ out, int total) {
    float m = 0.0f;
    for (int i = blockIdx.x * 256 + threadIdx.x; i < total; i += gridDim.x * 256)
        m = fmaxf(m, fabsf(tanhf(w[i])));
    __shared__ float ls[256];
    ls[threadIdx.x] = m;
    __syncthreads();
    for (int o = 128; o > 0; o >>= 1) {
        if (threadIdx.x < o) ls[threadIdx.x] = fmaxf(ls[threadIdx.x], ls[threadIdx.x + o]);
        __syncthreads();
    }
    if (threadIdx.x == 0) atomicMax(out, __float_as_uint(ls[0]));
}

__global__ __launch_bounds__(256)
void wquant_kernel(const float* __restrict__ w, const unsigned int* __restrict__ maxw_bits,
                   signed char* __restrict__ wq, int total) {
    int idx = blockIdx.x * 256 + threadIdx.x;
    if (idx >= total) return;
    int tap = idx % 9;
    int ci  = (idx / 9) & (C_CH - 1);
    int co  = idx / (9 * C_CH);
    float maxw = __uint_as_float(maxw_bits[0]);
    float t  = tanhf(w[idx]);
    float wn = t / (2.0f * maxw) + 0.5f;
    int   k  = (int)rintf(wn * 15.0f);
    wq[(tap * C_CH + co) * C_CH + ci] = (signed char)(2 * k - 15);
}

__global__ __launch_bounds__(256)
void bn_stats_kernel(const float* __restrict__ x, const float* __restrict__ gamma,
                     const float* __restrict__ beta, float* __restrict__ scaleA,
                     float* __restrict__ shiftB) {
    int c = blockIdx.x;
    int tid = threadIdx.x;
    float s = 0.0f, s2 = 0.0f;
    for (int n = 0; n < NIMG; ++n) {
        const float* p = x + (size_t)(n * C_CH + c) * HW;
        for (int i = tid; i < HW; i += 256) {
            float v = p[i];
            s += v;
            s2 += v * v;
        }
    }
    __shared__ float ls[256], ls2[256];
    ls[tid] = s; ls2[tid] = s2;
    __syncthreads();
    for (int o = 128; o > 0; o >>= 1) {
        if (tid < o) { ls[tid] += ls[tid + o]; ls2[tid] += ls2[tid + o]; }
        __syncthreads();
    }
    if (tid == 0) {
        const float cnt = (float)(NIMG * HW);
        float mean = ls[0] / cnt;
        float var  = ls2[0] / cnt - mean * mean;
        float a = gamma[c] * rsqrtf(var + 1e-5f);
        scaleA[c] = a;
        shiftB[c] = beta[c] - mean * a;
    }
}

__global__ __launch_bounds__(256)
void act_quant_nhwc_kernel(const float* __restrict__ x, const float* __restrict__ scaleA,
                           const float* __restrict__ shiftB, unsigned char* __restrict__ qT) {
    const int y = blockIdx.x;
    const int n = blockIdx.y;
    __shared__ unsigned char qt[C_CH * W_DIM];
    const int t = threadIdx.x;
    for (int f = t; f < C_CH * W_DIM; f += 256) {
        int c = f / W_DIM, xc = f - c * W_DIM;
        float v = x[((size_t)(n * C_CH + c) * H_DIM + y) * W_DIM + xc];
        float z = fminf(fmaxf(scaleA[c] * v + shiftB[c], 0.0f), 1.0f);
        qt[f] = (unsigned char)(int)rintf(z * 15.0f);
    }
    __syncthreads();
    for (int g = t; g < C_CH * W_DIM; g += 256) {
        int xc = g >> 8, c = g & (C_CH - 1);
        qT[(((size_t)n * H_DIM + y) * W_DIM + xc) * C_CH + c] = qt[c * W_DIM + xc];
    }
}

__global__ __launch_bounds__(256)
void conv3x3_q_wmma_kernel(const unsigned char* __restrict__ actT,
                           const signed char*  __restrict__ wq,
                           const float*        __restrict__ maxw,
                           const float*        __restrict__ shortcut,
                           float*              __restrict__ out,
                           int add_shortcut) {
    const int coutBase = blockIdx.x * 128;
    const int y = blockIdx.y;
    const int n = blockIdx.z;

    __shared__ __align__(16) signed char   Atile[9 * 128 * 64];
    __shared__ __align__(16) unsigned char Brows[3 * BPX * 64];

    const int tid   = threadIdx.x;
    const int lane  = tid & 31;
    const int wave  = tid >> 5;
    const int waveM = wave & 3;
    const int waveN = wave >> 2;
    const int hi    = (lane >> 4) & 1;
    const int l16   = lane & 15;

    v8i zero;
#pragma unroll
    for (int i = 0; i < 8; ++i) zero[i] = 0;
    v8i acc[2][2];
    acc[0][0] = zero; acc[0][1] = zero; acc[1][0] = zero; acc[1][1] = zero;

    for (int c0 = 0; c0 < C_CH; c0 += 64) {
        for (int f = tid; f < 3 * BPX * 4; f += 256) {
            int r   = f / (BPX * 4);
            int rem = f - r * (BPX * 4);
            int p   = rem >> 2;
            int q   = rem & 3;
            int ys  = y + r - 1;
            int xs  = p - 1;
            int4 v; v.x = 0; v.y = 0; v.z = 0; v.w = 0;
            if (ys >= 0 && ys < H_DIM && xs >= 0 && xs < W_DIM)
                v = *(const int4*)(actT + ((((size_t)n * H_DIM + ys) * W_DIM + xs) * C_CH + c0 + q * 16));
            *(int4*)(Brows + ((r * BPX + p) * 64) + q * 16) = v;
        }
        for (int rr = tid; rr < 9 * 128; rr += 256) {
            int tap = rr >> 7;
            int m   = rr & 127;
            const int4* src = (const int4*)(wq + ((size_t)(tap * C_CH + coutBase + m) * C_CH + c0));
            __builtin_prefetch(src + 16, 0, 1);
            int4* dst = (int4*)(Atile + (tap * 128 + m) * 64);
            dst[0] = src[0]; dst[1] = src[1]; dst[2] = src[2]; dst[3] = src[3];
        }
        __syncthreads();

        const int* Ai = (const int*)Atile;
        const int* Bi = (const int*)Brows;
#pragma unroll
        for (int dy = 0; dy < 3; ++dy) {
#pragma unroll
            for (int dx = 0; dx < 3; ++dx) {
                const int tap = dy * 3 + dx;
                v8i afrag[2], bfrag[2];
#pragma unroll
                for (int mt = 0; mt < 2; ++mt) {
                    int row = tap * 128 + waveM * 32 + mt * 16 + l16;
#pragma unroll
                    for (int v = 0; v < 8; ++v) {
                        int sub = v & 3;
                        int dwi = ((v & 4) ? 8 : 0) + sub + hi * 4;
                        afrag[mt][v] = Ai[row * 16 + dwi];
                    }
                }
#pragma unroll
                for (int nt = 0; nt < 2; ++nt) {
                    int p = waveN * 32 + nt * 16 + l16 + dx;
#pragma unroll
                    for (int v = 0; v < 8; ++v) {
                        int dwi = (v & 3) + ((v & 4) ? 8 : 0) + hi * 4;
                        bfrag[nt][v] = Bi[(dy * BPX + p) * 16 + dwi];
                    }
                }
#pragma unroll
                for (int mt = 0; mt < 2; ++mt)
#pragma unroll
                    for (int nt = 0; nt < 2; ++nt)
                        acc[mt][nt] = __builtin_amdgcn_wmma_i32_16x16x64_iu8(
                             true,  afrag[mt],
                             false, bfrag[nt],
                            acc[mt][nt],  false,  false);
            }
        }
        __syncthreads();
    }

    const float scale = maxw[0] * (1.0f / 225.0f);
#pragma unroll
    for (int mt = 0; mt < 2; ++mt) {
#pragma unroll
        for (int nt = 0; nt < 2; ++nt) {
            int px = waveN * 32 + nt * 16 + l16;
            if (px >= W_DIM) continue;
#pragma unroll
            for (int r = 0; r < 8; ++r) {
                int m  = r + hi * 8;
                int ch = coutBase + waveM * 32 + mt * 16 + m;
                size_t idx = ((size_t)(n * C_CH + ch) * H_DIM + y) * W_DIM + px;
                float v = (float)acc[mt][nt][r] * scale;
                if (add_shortcut) v += shortcut[idx];
                out[idx] = v;
            }
        }
    }
}

extern "C" void kernel_launch(void* const* d_in, const int* in_sizes, int n_in,
                              void* d_out, int out_size, void* d_ws, size_t ws_size,
                              hipStream_t stream) {
    const float* x          = (const float*)d_in[0];
    const float* bn0_gamma  = (const float*)d_in[1];
    const float* bn0_beta   = (const float*)d_in[2];
    const float* conv0_w    = (const float*)d_in[3];
    const float* bn1_gamma  = (const float*)d_in[4];
    const float* bn1_beta   = (const float*)d_in[5];
    const float* conv1_w    = (const float*)d_in[6];
    float* out = (float*)d_out;

    char* ws = (char*)d_ws;
    unsigned char* actT0 = (unsigned char*)(ws);
    unsigned char* actT1 = (unsigned char*)(ws + 12845056);
    float*         y0    = (float*)(ws + 25690112);
    signed char*   wq0   = (signed char*)(ws + 77070336);
    signed char*   wq1   = (signed char*)(ws + 77660160);
    float*         sA0   = (float*)(ws + 78249984);
    float*         sB0   = (float*)(ws + 78251008);
    float*         sA1   = (float*)(ws + 78252032);
    float*         sB1   = (float*)(ws + 78253056);
    unsigned int*  mw0   = (unsigned int*)(ws + 78254080);
    unsigned int*  mw1   = (unsigned int*)(ws + 78254084);

    hipMemsetAsync(ws + 78254080, 0, 8, stream);

    wmax_kernel<<<512, 256, 0, stream>>>(conv0_w, mw0, W_ELEMS);
    wmax_kernel<<<512, 256, 0, stream>>>(conv1_w, mw1, W_ELEMS);
    wquant_kernel<<<(W_ELEMS + 255) / 256, 256, 0, stream>>>(conv0_w, mw0, wq0, W_ELEMS);
    wquant_kernel<<<(W_ELEMS + 255) / 256, 256, 0, stream>>>(conv1_w, mw1, wq1, W_ELEMS);

    bn_stats_kernel<<<C_CH, 256, 0, stream>>>(x, bn0_gamma, bn0_beta, sA0, sB0);
    dim3 qgrid(H_DIM, NIMG);
    act_quant_nhwc_kernel<<<qgrid, 256, 0, stream>>>(x, sA0, sB0, actT0);

    dim3 cgrid(2, H_DIM, NIMG);
    conv3x3_q_wmma_kernel<<<cgrid, 256, 0, stream>>>(actT0, wq0, (const float*)mw0, x, y0, 0);

    bn_stats_kernel<<<C_CH, 256, 0, stream>>>(y0, bn1_gamma, bn1_beta, sA1, sB1);
    act_quant_nhwc_kernel<<<qgrid, 256, 0, stream>>>(y0, sA1, sB1, actT1);

    conv3x3_q_wmma_kernel<<<cgrid, 256, 0, stream>>>(actT1, wq1, (const float*)mw1, x, out, 1);
}
